// SSIMLoss_1898375545654
// MI455X (gfx1250) — matrix operand order repaired and checked by reading, not compiled, not run on hardware
//
#include <hip/hip_runtime.h>
#include <hip/hip_bf16.h>

// SSIM loss, MI455X (gfx1250, wave32).
// Separable 11-tap Gaussian conv expressed as band-matrix WMMA:
//   D(16x16) = A(16x32) x B(32x16),  A(i,k) = g[k-i] (Gaussian band)
// Same A fragment drives horizontal and vertical passes. f16 inputs, f32 acc.
// Input patches are filled with GLOBAL_LOAD_ASYNC_TO_LDS_B32 (ASYNCcnt path);
// zero-halo lanes fall back to direct DS stores.

typedef __attribute__((ext_vector_type(16))) _Float16 v16h;
typedef __attribute__((ext_vector_type(8)))  float    v8f;

#define IMG_H 512
#define IMG_W 512
#define NIMG  64
#define ROWS_PER_BLK 128      // 8 waves x 16 rows
#define COLS_PER_BLK 16
#define HALO 5
#define PATCH_ROWS 144        // 9 groups of 16 rows (covers 128 + 2*5 halo, padded)
#define RAWS 33               // raw patch LDS row stride (floats), odd to spread banks
#define HS   18               // h-buffer LDS row stride (f16), odd-ish to spread banks
#define SSIM_C1 1.0e-4f
#define SSIM_C2 9.0e-4f

__global__ __launch_bounds__(256) void ssim_tile_kernel(
    const float* __restrict__ pred, const float* __restrict__ gt,
    float* __restrict__ partial)
{
    __shared__ float    rawP[PATCH_ROWS * RAWS];
    __shared__ float    rawG[PATCH_ROWS * RAWS];
    __shared__ _Float16 hbuf[PATCH_ROWS * HS];
    __shared__ float    red[256];

    const int tid  = threadIdx.x;
    const int lane = tid & 31;
    const int wv   = tid >> 5;                 // 0..7
    const int colbase = blockIdx.x * COLS_PER_BLK;
    const int rowbase = blockIdx.y * ROWS_PER_BLK;
    const size_t ibase = (size_t)blockIdx.z * IMG_H * IMG_W;

    // ---- async fill of 144x32 halo patch: DMA straight into LDS ----
    // In-range lanes: global_load_async_to_lds_b32 (ASYNCcnt, no VGPR staging).
    // Out-of-range (zero-pad halo) lanes: plain LDS zero store.
    for (int idx = tid; idx < PATCH_ROWS * 32; idx += 256) {
        int pr = idx >> 5;
        int pc = idx & 31;
        int ar = rowbase + pr - HALO;
        int ac = colbase + pc - HALO;
        bool ok = (ar >= 0) & (ar < IMG_H) & (ac >= 0) & (ac < IMG_W);
        int li = pr * RAWS + pc;
        if (ok) {
            size_t off = ibase + (size_t)ar * IMG_W + ac;
            unsigned ldsP = (unsigned)(size_t)(&rawP[li]);   // low 32 bits = LDS offset
            unsigned ldsG = (unsigned)(size_t)(&rawG[li]);
            const float* gp = pred + off;
            const float* gq = gt + off;
            asm volatile("global_load_async_to_lds_b32 %0, %1, off"
                         :: "v"(ldsP), "v"(gp) : "memory");
            asm volatile("global_load_async_to_lds_b32 %0, %1, off"
                         :: "v"(ldsG), "v"(gq) : "memory");
        } else {
            rawP[li] = 0.0f;
            rawG[li] = 0.0f;
        }
    }

    // ---- Gaussian band A fragment: A(m,k) = g[k-m], g = normalized sigma=11/6 ----
    // (computed while the async fill is in flight)
    // A k order used here (contiguous, same as B): lanes 0-15 hold K=0..15; lanes 16-31 hold K=16..31
    const int halfA = lane >> 4;
    const int m     = lane & 15;
    v16h afrag;
#pragma unroll
    for (int e = 0; e < 16; ++e) {
        int k = halfA * 16 + e;
        int t = k - m;                              // tap index 0..10 inside band
        float d = (float)(t - 5);
        float wgt = 0.2181254f * exp2f(-0.2146206f * d * d);  // g[t], pre-normalized
        afrag[e] = (_Float16)(((t >= 0) && (t < 11)) ? wgt : 0.0f);
    }

    asm volatile("s_wait_asynccnt 0x0" ::: "memory");   // async LDS fill complete (this wave)
    __syncthreads();                                    // visible block-wide

    v8f acc[5];
#pragma unroll
    for (int c = 0; c < 5; ++c) {
        // ---- stage 1: horizontal conv, one 16-row group per wave (wave7 does 2) ----
        auto stage1 = [&](int grp) {
            int j  = lane & 15;                     // row within group (N dim)
            int hb = lane >> 4;
            int pr = 16 * grp + j;                  // raw patch row
            v16h bfrag;
#pragma unroll
            for (int e = 0; e < 16; ++e) {
                int K = hb * 16 + e;                // input column (K dim), 0..31
                float p = rawP[pr * RAWS + K];
                float q = rawG[pr * RAWS + K];
                float v = (c == 0) ? p : (c == 1) ? q
                        : (c == 2) ? p * p : (c == 3) ? q * q : p * q;
                bfrag[e] = (_Float16)v;             // K>=26 garbage is killed by A=0
            }
            v8f z = {};
            v8f hres = __builtin_amdgcn_wmma_f32_16x16x32_f16(
                false, afrag, false, bfrag, (short)0, z, false, false);
            // D layout: VGPR r -> M = r + 8*(lane>=16), N = lane&15
            int N  = lane & 15;
            int hd = lane >> 4;
#pragma unroll
            for (int r = 0; r < 8; ++r)
                hbuf[(16 * grp + N) * HS + (r + 8 * hd)] = (_Float16)hres[r];
        };
        stage1(wv);
        if (wv == 7) stage1(8);                     // wave-uniform branch: EXEC stays all-1s
        __syncthreads();

        // ---- stage 2: vertical conv, one 16x16 output tile per wave ----
        {
            int j2 = lane & 15;                     // output column (N dim)
            int hb = lane >> 4;
            v16h b2;
#pragma unroll
            for (int e = 0; e < 16; ++e) {
                int K = hb * 16 + e;                // input row (K dim), 0..31
                b2[e] = hbuf[(16 * wv + K) * HS + j2];   // rows <=143 always valid
            }
            v8f z = {};
            acc[c] = __builtin_amdgcn_wmma_f32_16x16x32_f16(
                false, afrag, false, b2, (short)0, z, false, false);
        }
        __syncthreads();                            // hbuf reused next channel
    }

    // ---- fused pointwise SSIM + block reduction ----
    float local = 0.0f;
#pragma unroll
    for (int r = 0; r < 8; ++r) {
        float mu1 = acc[0][r], mu2 = acc[1][r];
        float s1  = acc[2][r] - mu1 * mu1;
        float s2  = acc[3][r] - mu2 * mu2;
        float s12 = acc[4][r] - mu1 * mu2;
        float num = (2.0f * mu1 * mu2 + SSIM_C1) * (2.0f * s12 + SSIM_C2);
        float den = (mu1 * mu1 + mu2 * mu2 + SSIM_C1) * (s1 + s2 + SSIM_C2);
        local += num / den;
    }
    red[tid] = local;
    __syncthreads();
#pragma unroll
    for (int s = 128; s > 0; s >>= 1) {
        if (tid < s) red[tid] += red[tid + s];
        __syncthreads();
    }
    if (tid == 0)
        partial[(blockIdx.z * gridDim.y + blockIdx.y) * gridDim.x + blockIdx.x] = red[0];
}

__global__ __launch_bounds__(256) void ssim_reduce_kernel(
    const float* __restrict__ partial, float* __restrict__ out, int nPartial)
{
    __shared__ float red[256];
    float local = 0.0f;
    for (int i = threadIdx.x; i < nPartial; i += 256) local += partial[i];
    red[threadIdx.x] = local;
    __syncthreads();
#pragma unroll
    for (int s = 128; s > 0; s >>= 1) {
        if (threadIdx.x < s) red[threadIdx.x] += red[threadIdx.x + s];
        __syncthreads();
    }
    if (threadIdx.x == 0)
        out[0] = 1.0f - red[0] * (1.0f / 16777216.0f);   // / (64*512*512)
}

extern "C" void kernel_launch(void* const* d_in, const int* in_sizes, int n_in,
                              void* d_out, int out_size, void* d_ws, size_t ws_size,
                              hipStream_t stream) {
    const float* pred = (const float*)d_in[0];
    const float* gt   = (const float*)d_in[1];
    float* partials   = (float*)d_ws;              // 8192 floats, fully rewritten each call

    dim3 grid(IMG_W / COLS_PER_BLK, IMG_H / ROWS_PER_BLK, NIMG);   // 32 x 4 x 64
    ssim_tile_kernel<<<grid, 256, 0, stream>>>(pred, gt, partials);

    int nPartial = grid.x * grid.y * grid.z;
    ssim_reduce_kernel<<<1, 256, 0, stream>>>(partials, (float*)d_out, nPartial);
}
